// DLFGO_twoplane_70471823393083
// MI455X (gfx1250) — matrix operand order repaired and checked by reading, not compiled, not run on hardware
//
#include <hip/hip_runtime.h>
#include <hip/hip_bf16.h>
#include <math.h>

typedef _Float16 v8h  __attribute__((ext_vector_type(8)));
typedef _Float16 v16h __attribute__((ext_vector_type(16)));
typedef float    v8f  __attribute__((ext_vector_type(8)));

#define N_RAYS          262144
#define GRID_C          16
#define MLP_DIM         288      // 3 levels * 6 planes * 16 channels
#define MLP_W           128
#define RAYS_PER_BLOCK  64
#define THREADS         128      // 4 wave32
#define FEAT_STRIDE_B   576      // 288 halves (feats) >= 128 floats (h) per ray row

// ---------------------------------------------------------------------------
// Kernel 1: pack w1 (f32 [288][128], K-major) into f16 WMMA B-fragment order.
// B layout (32x16 tile), matching A: lanes 0-15 hold K {0..7},{16..23} at N=lane,
// lanes 16-31 hold K {8..15},{24..31} at N=lane-16.
// Fragment (nt,kk) covers N in [16*nt,16*nt+16), K in [32*kk,32*kk+32).
// Each lane's 16 halves stored contiguously -> one 32B load in the GEMM.
// ---------------------------------------------------------------------------
__global__ void __launch_bounds__(256)
pack_w1_kernel(const float* __restrict__ w1, _Float16* __restrict__ wpk) {
    int idx = blockIdx.x * blockDim.x + threadIdx.x;   // 0..36863
    int K = idx >> 7;          // 0..287
    int N = idx & 127;         // 0..127
    int kk   = K >> 5;                                 // 0..8
    int nt   = N >> 4;                                 // 0..7
    int lane = (N & 15) | (((K >> 3) & 1) << 4);       // 0..31
    int h16  = (K & 7) | (((K >> 4) & 1) << 3);        // position in lane chunk
    wpk[(((nt * 9 + kk) * 32) + lane) * 16 + h16] = (_Float16)w1[K * 128 + N];
}

__device__ __forceinline__ v16h load_afrag(const char* frow, int kk, int ko0) {
    const _Float16* fp = (const _Float16*)frow + 32 * kk + ko0;
    v8h lo = *(const v8h*)fp;
    v8h hi = *(const v8h*)(fp + 16);
    return __builtin_shufflevector(lo, hi,
                                   0, 1, 2, 3, 4, 5, 6, 7,
                                   8, 9, 10, 11, 12, 13, 14, 15);
}

// ---------------------------------------------------------------------------
// Kernel 2: fused feature-gather + MLP.
// ---------------------------------------------------------------------------
__global__ void __launch_bounds__(THREADS)
twoplane_fused_kernel(const float* __restrict__ ray,
                      const float* __restrict__ g0,
                      const float* __restrict__ g1,
                      const float* __restrict__ g2,
                      const _Float16* __restrict__ wpk,
                      const float* __restrict__ b1,
                      const float* __restrict__ w2,
                      const float* __restrict__ b2,
                      float* __restrict__ out) {
    extern __shared__ char sFB[];            // 64 rows x 576B: feats f16, then h f32
    __shared__ float sB1[MLP_W];
    __shared__ float sW2[MLP_W * 3];

    const int tid = threadIdx.x;
    if (tid < MLP_W) sB1[tid] = b1[tid];
    for (int i = tid; i < MLP_W * 3; i += THREADS) sW2[i] = w2[i];

    // ---- Phase 1: bilinear feature gather (64 rays x 18 plane-levels) ----
    for (int t = tid; t < RAYS_PER_BLOCK * 18; t += THREADS) {
        int rloc  = t / 18;
        int pl    = t - rloc * 18;
        int level = pl / 6;
        int comb  = pl - level * 6;
        int rg    = blockIdx.x * RAYS_PER_BLOCK + rloc;

        // COMBS = {0,1},{0,2},{0,3},{1,2},{1,3},{2,3} nibble-packed
        int a0 = (0x211000 >> (comb * 4)) & 0xF;
        int a1 = (0x332321 >> (comb * 4)) & 0xF;
        float p0 = ray[rg * 4 + a0];
        float p1 = ray[rg * 4 + a1];

        const float* g; int H;
        if (level == 0)      { g = g0; H = 128; }
        else if (level == 1) { g = g1; H = 256; }
        else                 { g = g2; H = 512; }

        float u  = p0 * (float)(H - 1);
        float v  = p1 * (float)(H - 1);
        float u0 = fminf(fmaxf(floorf(u), 0.f), (float)(H - 2));
        float v0 = fminf(fmaxf(floorf(v), 0.f), (float)(H - 2));
        float wu = u - u0;
        float wv = v - v0;
        float w00 = (1.f - wu) * (1.f - wv);
        float w01 = (1.f - wu) * wv;
        float w10 = wu * (1.f - wv);
        float w11 = wu * wv;

        int HW = H * H;
        const float* base = g + comb * (GRID_C * HW) + (int)u0 * H + (int)v0;

        _Float16 fq[16];
        #pragma unroll
        for (int c = 0; c < GRID_C; ++c) {
            const float* p = base + c * HW;
            float s = p[0] * w00 + p[1] * w01 + p[H] * w10 + p[H + 1] * w11;
            fq[c] = (_Float16)s;
        }
        // feature index = level*96 + comb*16 + c   (matches reference ordering)
        v8h* dst = (v8h*)(sFB + rloc * FEAT_STRIDE_B + (level * 96 + comb * 16) * 2);
        dst[0] = *(const v8h*)&fq[0];
        dst[1] = *(const v8h*)&fq[8];
    }
    __syncthreads();

    // ---- Phase 2: layer 1 GEMM via WMMA (each wave: 16 rays) ----
    const int lane    = tid & 31;
    const int wave    = tid >> 5;
    const int rayBase = wave * 16;

    // A fragment addressing (16-bit A 16x32 layout):
    // lanes 0-15: row=lane,    K chunks {0..7},{16..23}
    // lanes16-31: row=lane-16, K chunks {8..15},{24..31}
    const int rowA = lane & 15;
    const int ko0  = (lane >> 4) << 3;
    const char* frow = sFB + (rayBase + rowA) * FEAT_STRIDE_B;
    const v16h* bptr = (const v16h*)wpk;

    v8f zero = {0.f, 0.f, 0.f, 0.f, 0.f, 0.f, 0.f, 0.f};
    v8f acc[8];
    #pragma unroll
    for (int nt = 0; nt < 8; ++nt) acc[nt] = zero;

    // Software-pipelined streams:
    //  - B fragments: 4-deep prefetch ring (frag i consumed, frag i+4 loading)
    //  - A fragments: double-buffered across k-steps
    // Iteration order i = kk*8 + nt; fragment id in wpk = nt*9 + kk.
    v16h bbuf[4];
    #pragma unroll
    for (int j = 0; j < 4; ++j) {
        int kkj = j >> 3, ntj = j & 7;
        bbuf[j] = bptr[(ntj * 9 + kkj) * 32 + lane];
    }
    v16h acur = load_afrag(frow, 0, ko0);
    v16h anext;

    #pragma unroll
    for (int i = 0; i < 72; ++i) {
        const int kk = i >> 3;
        const int nt = i & 7;
        if (nt == 0 && kk < 8) anext = load_afrag(frow, kk + 1, ko0);

        v16h bcur = bbuf[i & 3];
        const int j = i + 4;
        if (j < 72) {
            int kkj = j >> 3, ntj = j & 7;
            bbuf[i & 3] = bptr[(ntj * 9 + kkj) * 32 + lane];
        }

        acc[nt] = __builtin_amdgcn_wmma_f32_16x16x32_f16(
            false, acur, false, bcur, (short)0, acc[nt], false, false);

        if (nt == 7 && kk < 8) acur = anext;
    }

    // ---- bias + ReLU; scatter h into LDS rows (C/D layout) ----
    // element (M,N): VGPR r -> M = r + 8*(lane>=16), N = 16*nt + (lane&15)
    const int ncol = lane & 15;
    const int mhi  = (lane >> 4) << 3;
    #pragma unroll
    for (int nt = 0; nt < 8; ++nt) {
        int N = nt * 16 + ncol;
        float bv = sB1[N];
        #pragma unroll
        for (int r = 0; r < 8; ++r) {
            float hval = acc[nt][r] + bv;
            hval = fmaxf(hval, 0.f);
            float* hrow = (float*)(sFB + (rayBase + mhi + r) * FEAT_STRIDE_B);
            hrow[N] = hval;
        }
    }
    __syncthreads();

    // ---- Phase 3: layer 2 (128 -> 3) + sigmoid, one thread per ray ----
    if (tid < RAYS_PER_BLOCK) {
        int rg = blockIdx.x * RAYS_PER_BLOCK + tid;
        const float* hrow = (const float*)(sFB + tid * FEAT_STRIDE_B);
        float o0 = b2[0], o1 = b2[1], o2 = b2[2];
        #pragma unroll 8
        for (int k = 0; k < MLP_W; ++k) {
            float hv = hrow[k];
            o0 += hv * sW2[k * 3 + 0];
            o1 += hv * sW2[k * 3 + 1];
            o2 += hv * sW2[k * 3 + 2];
        }
        out[rg * 3 + 0] = 1.f / (1.f + expf(-o0));
        out[rg * 3 + 1] = 1.f / (1.f + expf(-o1));
        out[rg * 3 + 2] = 1.f / (1.f + expf(-o2));
    }
}

extern "C" void kernel_launch(void* const* d_in, const int* in_sizes, int n_in,
                              void* d_out, int out_size, void* d_ws, size_t ws_size,
                              hipStream_t stream) {
    const float* ray = (const float*)d_in[0];
    const float* g0  = (const float*)d_in[1];   // [6,16,128,128]
    const float* g1  = (const float*)d_in[2];   // [6,16,256,256]
    const float* g2  = (const float*)d_in[3];   // [6,16,512,512]
    const float* w1  = (const float*)d_in[4];   // [288,128]
    const float* b1  = (const float*)d_in[5];   // [128]
    const float* w2  = (const float*)d_in[6];   // [128,3]
    const float* b2  = (const float*)d_in[7];   // [3]
    float* out = (float*)d_out;                 // [262144,3]

    _Float16* wpk = (_Float16*)d_ws;            // 36864 halves = 73728 B scratch

    pack_w1_kernel<<<(MLP_DIM * MLP_W) / 256, 256, 0, stream>>>(w1, wpk);

    twoplane_fused_kernel<<<N_RAYS / RAYS_PER_BLOCK, THREADS,
                            RAYS_PER_BLOCK * FEAT_STRIDE_B, stream>>>(
        ray, g0, g1, g2, wpk, b1, w2, b2, out);
}
